// GroupedExperts_58772332479042
// MI455X (gfx1250) — matrix operand order repaired and checked by reading, not compiled, not run on hardware
//
#include <hip/hip_runtime.h>
#include <math.h>

typedef __attribute__((ext_vector_type(16))) __bf16 v16bf;
typedef __attribute__((ext_vector_type(8)))  float  v8f;

#define T_TOK  16384
#define DMODEL 2048
#define NEXP   8
#define HIDN   5632
#define GTOK   2048   // tokens per expert (T/E)
#define LDK    72     // padded LDS stride for BK=64 tiles (144 B rows, 16B aligned)

// ---------------------------------------------------------------------------
// Async global->LDS staging (CDNA5 GLOBAL_LOAD_ASYNC_TO_LDS_B128, ASYNCcnt),
// guarded so we fall back to load+ds_store if the builtin is unavailable.
// Builtin signature (from hipcc diagnostic): params are pointers to a 16-byte
// int vector; arg0 = global (AS1) source, arg1 = LDS (AS3) dest, then
// imm offset and imm cpol.
// ---------------------------------------------------------------------------
#if __has_builtin(__builtin_amdgcn_global_load_async_to_lds_b128) && \
    __has_builtin(__builtin_amdgcn_s_wait_asynccnt)
#define USE_ASYNC_LDS 1
#else
#define USE_ASYNC_LDS 0
#endif

typedef int v4i_t __attribute__((vector_size(16)));

__device__ __forceinline__ void copy16(unsigned short* lds, const unsigned short* g) {
#if USE_ASYNC_LDS
  __builtin_amdgcn_global_load_async_to_lds_b128(
      (__attribute__((address_space(1))) v4i_t*)(g),
      (__attribute__((address_space(3))) v4i_t*)(lds), 0, 0);
#else
  *(uint4*)lds = *(const uint4*)g;
#endif
}

__device__ __forceinline__ void wait_stage() {
#if USE_ASYNC_LDS
  __builtin_amdgcn_s_wait_asynccnt(0);
#endif
}

// ---------------------------------------------------------------------------
// f32 -> bf16 helpers (round-to-nearest-even); packed cvt if available
// ---------------------------------------------------------------------------
__device__ __forceinline__ unsigned short f2bf(float x) {
  unsigned u = __float_as_uint(x);
  u += 0x7FFFu + ((u >> 16) & 1u);
  return (unsigned short)(u >> 16);
}

__device__ __forceinline__ unsigned pack2bf(float a, float b) {
#if __has_builtin(__builtin_amdgcn_cvt_pk_bf16_f32)
  typedef __attribute__((ext_vector_type(2))) __bf16 v2bf;
  union { v2bf v; unsigned u; } cu;
  cu.v = __builtin_amdgcn_cvt_pk_bf16_f32(a, b);
  return cu.u;
#else
  return (unsigned)f2bf(a) | ((unsigned)f2bf(b) << 16);
#endif
}

__device__ __forceinline__ float fast_rcp(float x) {
#if __has_builtin(__builtin_amdgcn_rcpf)
  return __builtin_amdgcn_rcpf(x);
#else
  return 1.0f / x;
#endif
}

// ---------------------------------------------------------------------------
// WMMA fragment loaders per CDNA5 ISA VGPR layouts.
// A 16x32 bf16 (LDS row-major [m][k], stride LDK): half h=lane>>4, m=lane&15;
//   j0..j7 = K [koff+h*8, +8), j8..j15 = K [koff+16+h*8, +8)
// B 32x16 bf16 (LDS K-major [n][k]): n=lane&15; j0..j7 = K [koff+h*8, +8), j8..j15 = K [koff+16+h*8, +8)
// ---------------------------------------------------------------------------
__device__ __forceinline__ v16bf ld_a_frag(const unsigned short* As, int mrow, int koff,
                                           int lane) {
  const int half = lane >> 4;
  const int m = (lane & 15) + mrow;
  const unsigned short* p = As + m * LDK + koff + half * 8;
  union { v16bf v; uint4 q[2]; } u;
  u.q[0] = *(const uint4*)(p);
  u.q[1] = *(const uint4*)(p + 16);
  return u.v;
}

__device__ __forceinline__ v16bf ld_b_frag(const unsigned short* Bs, int ncol, int koff,
                                           int lane) {
  const int half = lane >> 4;
  const int n = (lane & 15) + ncol;
  const unsigned short* p = Bs + n * LDK + koff + half * 8;
  union { v16bf v; uint4 q[2]; } u;
  u.q[0] = *(const uint4*)(p);
  u.q[1] = *(const uint4*)(p + 16);
  return u.v;
}

#define WMMA_BF16(A, B, C) \
  __builtin_amdgcn_wmma_f32_16x16x32_bf16(false, (A), false, (B), (short)0, (C), false, false)

// ---------------------------------------------------------------------------
// Prepass A: x f32 [T,D] -> bf16 [T,D]
// ---------------------------------------------------------------------------
__global__ void __launch_bounds__(256)
convert_x_kernel(const float* __restrict__ in, unsigned short* __restrict__ out) {
  const size_t f = (size_t)blockIdx.x * 256 + threadIdx.x;  // float4 index
  const float4 v = *(const float4*)(in + f * 4);
  uint2 pk;
  pk.x = pack2bf(v.x, v.y);
  pk.y = pack2bf(v.z, v.w);
  *(uint2*)(out + f * 4) = pk;
}

// ---------------------------------------------------------------------------
// Prepass B: per-expert transpose + convert: in f32 [R,C] -> out bf16 [C,R]
// ---------------------------------------------------------------------------
__global__ void __launch_bounds__(256)
transpose_convert_kernel(const float* __restrict__ in, unsigned short* __restrict__ out,
                         int R, int C) {
  __shared__ unsigned short t[64][72];
  const size_t mat = (size_t)blockIdx.z * R * C;
  const float* inp = in + mat;
  unsigned short* outp = out + mat;
  const int rb = blockIdx.y * 64, cb = blockIdx.x * 64;
  const int tid = threadIdx.x;
#pragma unroll
  for (int i = 0; i < 4; ++i) {
    const int f = tid + i * 256;        // 0..1023
    const int r = f >> 4, c4 = (f & 15) * 4;
    const float4 v = *(const float4*)(inp + (size_t)(rb + r) * C + cb + c4);
    t[r][c4 + 0] = f2bf(v.x);
    t[r][c4 + 1] = f2bf(v.y);
    t[r][c4 + 2] = f2bf(v.z);
    t[r][c4 + 3] = f2bf(v.w);
  }
  __syncthreads();
#pragma unroll
  for (int i = 0; i < 2; ++i) {
    const int f = tid + i * 256;        // 0..511
    const int c = f >> 3, r8 = (f & 7) * 8;
    unsigned tmp[8];
#pragma unroll
    for (int j = 0; j < 8; ++j) tmp[j] = t[r8 + j][c];
    uint4 pk;
    pk.x = tmp[0] | (tmp[1] << 16);
    pk.y = tmp[2] | (tmp[3] << 16);
    pk.z = tmp[4] | (tmp[5] << 16);
    pk.w = tmp[6] | (tmp[7] << 16);
    *(uint4*)(outp + (size_t)(cb + c) * R + rb + r8) = pk;
  }
}

// ---------------------------------------------------------------------------
// Kernel 1: h = silu(x@Wg) * (x@Wu); bf16 operands, weights K-major.
// Block tile 128 x (64+64), BK=64 (2 WMMA K-halves per stage), 8 waves (4x2),
// wave tile 32x32 per gate side -> 16 WMMAs/wave/stage. Double-buffered LDS,
// async global->LDS staging.
// ---------------------------------------------------------------------------
__global__ void __launch_bounds__(256)
moe_gemm1_silu_kernel(const unsigned short* __restrict__ xb,
                      const unsigned short* __restrict__ w13t,
                      unsigned short* __restrict__ h) {
  __shared__ unsigned short As[2][128 * LDK];
  __shared__ unsigned short B1[2][64 * LDK];
  __shared__ unsigned short B3[2][64 * LDK];

  const int nb  = blockIdx.x;   // 0..HID/64-1
  const int mb  = blockIdx.y;   // 0..GTOK/128-1
  const int e   = blockIdx.z;
  const int tid = threadIdx.x;
  const int lane = tid & 31;
  const int wid  = tid >> 5;
  const int wm = (wid >> 1) * 32;
  const int wn = (wid & 1) * 32;

  const unsigned short* xg = xb + (size_t)(e * GTOK + mb * 128) * DMODEL;
  const unsigned short* wg = w13t + ((size_t)e * (2 * HIDN) + nb * 64) * DMODEL;
  const unsigned short* wu = wg + (size_t)HIDN * DMODEL;

  auto stage = [&](int buf, int kk) {
    // A: 128 x 64 bf16 = 1024 x b128, 4 per thread
#pragma unroll
    for (int i = 0; i < 4; ++i) {
      const int f = tid + i * 256;
      const int row = f >> 3, c8 = (f & 7) * 8;
      copy16(&As[buf][row * LDK + c8], xg + (size_t)row * DMODEL + kk + c8);
    }
    // B gate/up: 64 x 64 bf16 = 512 x b128 each, 2 per thread each
#pragma unroll
    for (int i = 0; i < 2; ++i) {
      const int f = tid + i * 256;
      const int n = f >> 3, c8 = (f & 7) * 8;
      copy16(&B1[buf][n * LDK + c8], wg + (size_t)n * DMODEL + kk + c8);
      copy16(&B3[buf][n * LDK + c8], wu + (size_t)n * DMODEL + kk + c8);
    }
  };

  v8f acc1[2][2] = {};
  v8f acc3[2][2] = {};

  stage(0, 0);
  wait_stage();
  __syncthreads();

  for (int kk = 0; kk < DMODEL; kk += 64) {
    const int cur = (kk >> 6) & 1;
    if (kk + 64 < DMODEL) stage(cur ^ 1, kk + 64);

#pragma unroll
    for (int kh = 0; kh < 2; ++kh) {
      const int ko = kh * 32;
      const v16bf a0  = ld_a_frag(As[cur], wm + 0,  ko, lane);
      const v16bf a1  = ld_a_frag(As[cur], wm + 16, ko, lane);
      const v16bf b10 = ld_b_frag(B1[cur], wn + 0,  ko, lane);
      const v16bf b11 = ld_b_frag(B1[cur], wn + 16, ko, lane);
      const v16bf b30 = ld_b_frag(B3[cur], wn + 0,  ko, lane);
      const v16bf b31 = ld_b_frag(B3[cur], wn + 16, ko, lane);

      acc1[0][0] = WMMA_BF16(a0, b10, acc1[0][0]);
      acc1[0][1] = WMMA_BF16(a0, b11, acc1[0][1]);
      acc1[1][0] = WMMA_BF16(a1, b10, acc1[1][0]);
      acc1[1][1] = WMMA_BF16(a1, b11, acc1[1][1]);
      acc3[0][0] = WMMA_BF16(a0, b30, acc3[0][0]);
      acc3[0][1] = WMMA_BF16(a0, b31, acc3[0][1]);
      acc3[1][0] = WMMA_BF16(a1, b30, acc3[1][0]);
      acc3[1][1] = WMMA_BF16(a1, b31, acc3[1][1]);
    }
    wait_stage();
    __syncthreads();
  }

  // epilogue: h = silu(acc1) * acc3 (fast v_rcp sigmoid), write bf16
  const int half = lane >> 4;
  const int nlo  = lane & 15;
#pragma unroll
  for (int im = 0; im < 2; ++im) {
#pragma unroll
    for (int in_ = 0; in_ < 2; ++in_) {
#pragma unroll
      for (int r = 0; r < 8; ++r) {
        const float a = acc1[im][in_][r];
        const float b = acc3[im][in_][r];
        const float hv = a * fast_rcp(1.0f + __expf(-a)) * b;
        const int row = e * GTOK + mb * 128 + wm + im * 16 + half * 8 + r;
        const int col = nb * 64 + wn + in_ * 16 + nlo;
        h[(size_t)row * HIDN + col] = f2bf(hv);
      }
    }
  }
}

// ---------------------------------------------------------------------------
// Kernel 2: out = h(bf16) @ w2t(bf16, K-major [e][D][HID]), f32 store.
// Block tile 128x128, BK=64, 8 waves (4x2), wave tile 32x64 -> 16 WMMAs/stage.
// ---------------------------------------------------------------------------
__global__ void __launch_bounds__(256)
moe_gemm2_kernel(const unsigned short* __restrict__ h,
                 const unsigned short* __restrict__ w2t,
                 float* __restrict__ out) {
  __shared__ unsigned short As[2][128 * LDK];
  __shared__ unsigned short Bs[2][128 * LDK];

  const int nb  = blockIdx.x;   // 0..D/128-1
  const int mb  = blockIdx.y;
  const int e   = blockIdx.z;
  const int tid = threadIdx.x;
  const int lane = tid & 31;
  const int wid  = tid >> 5;
  const int wm = (wid >> 1) * 32;
  const int wn = (wid & 1) * 64;

  const unsigned short* hg = h + (size_t)(e * GTOK + mb * 128) * HIDN;
  const unsigned short* wt = w2t + ((size_t)e * DMODEL + nb * 128) * HIDN;
  const int colb = nb * 128;

  auto stage = [&](int buf, int kk) {
#pragma unroll
    for (int i = 0; i < 4; ++i) {
      const int f = tid + i * 256;
      const int row = f >> 3, c8 = (f & 7) * 8;
      copy16(&As[buf][row * LDK + c8], hg + (size_t)row * HIDN + kk + c8);
      copy16(&Bs[buf][row * LDK + c8], wt + (size_t)row * HIDN + kk + c8);
    }
  };

  v8f acc[2][4] = {};

  stage(0, 0);
  wait_stage();
  __syncthreads();

  for (int kk = 0; kk < HIDN; kk += 64) {
    const int cur = (kk >> 6) & 1;
    if (kk + 64 < HIDN) stage(cur ^ 1, kk + 64);

#pragma unroll
    for (int kh = 0; kh < 2; ++kh) {
      const int ko = kh * 32;
      const v16bf a0 = ld_a_frag(As[cur], wm + 0,  ko, lane);
      const v16bf a1 = ld_a_frag(As[cur], wm + 16, ko, lane);
#pragma unroll
      for (int t = 0; t < 4; ++t) {
        const v16bf bt = ld_b_frag(Bs[cur], wn + t * 16, ko, lane);
        acc[0][t] = WMMA_BF16(a0, bt, acc[0][t]);
        acc[1][t] = WMMA_BF16(a1, bt, acc[1][t]);
      }
    }
    wait_stage();
    __syncthreads();
  }

  const int half = lane >> 4;
  const int nlo  = lane & 15;
#pragma unroll
  for (int im = 0; im < 2; ++im) {
#pragma unroll
    for (int t = 0; t < 4; ++t) {
#pragma unroll
      for (int r = 0; r < 8; ++r) {
        const int row = e * GTOK + mb * 128 + wm + im * 16 + half * 8 + r;
        const int col = colb + wn + t * 16 + nlo;
        out[(size_t)row * DMODEL + col] = acc[im][t][r];
      }
    }
  }
}

// ---------------------------------------------------------------------------
// Kernel 3: expert_avg_top_scores[e] = mean(top_scores[e*g:(e+1)*g])
// ---------------------------------------------------------------------------
__global__ void __launch_bounds__(256)
moe_scores_kernel(const float* __restrict__ ts, float* __restrict__ out) {
  __shared__ float red[256];
  const int e = blockIdx.x;
  float s = 0.0f;
  for (int i = threadIdx.x; i < GTOK; i += 256) s += ts[e * GTOK + i];
  red[threadIdx.x] = s;
  __syncthreads();
  for (int off = 128; off > 0; off >>= 1) {
    if (threadIdx.x < off) red[threadIdx.x] += red[threadIdx.x + off];
    __syncthreads();
  }
  if (threadIdx.x == 0)
    out[(size_t)T_TOK * DMODEL + e] = red[0] * (1.0f / (float)GTOK);
}

// ---------------------------------------------------------------------------
// Launch. Inputs: (x, num_tokens_per_expert, top_scores, w13, w2).
// ws layout (bf16): h [T,HID] | x [T,D] | w13^T [E,2H,D] | w2^T [E,D,HID]
// ---------------------------------------------------------------------------
extern "C" void kernel_launch(void* const* d_in, const int* in_sizes, int n_in,
                              void* d_out, int out_size, void* d_ws, size_t ws_size,
                              hipStream_t stream) {
  (void)in_sizes; (void)n_in; (void)out_size; (void)ws_size;
  const float* x   = (const float*)d_in[0];
  const float* ts  = (const float*)d_in[2];
  const float* w13 = (const float*)d_in[3];
  const float* w2  = (const float*)d_in[4];
  float* out = (float*)d_out;

  unsigned short* hbuf = (unsigned short*)d_ws;
  unsigned short* xbf  = hbuf + (size_t)T_TOK * HIDN;
  unsigned short* w13t = xbf  + (size_t)T_TOK * DMODEL;
  unsigned short* w2t  = w13t + (size_t)NEXP * 2 * HIDN * DMODEL;

  dim3 blk(256);

  hipLaunchKernelGGL(convert_x_kernel, dim3((T_TOK * DMODEL / 4) / 256), blk, 0, stream,
                     x, xbf);
  hipLaunchKernelGGL(transpose_convert_kernel, dim3(2 * HIDN / 64, DMODEL / 64, NEXP),
                     blk, 0, stream, w13, w13t, DMODEL, 2 * HIDN);
  hipLaunchKernelGGL(transpose_convert_kernel, dim3(DMODEL / 64, HIDN / 64, NEXP),
                     blk, 0, stream, w2, w2t, HIDN, DMODEL);

  dim3 g1(HIDN / 64, GTOK / 128, NEXP);     // 88 x 16 x 8
  hipLaunchKernelGGL(moe_gemm1_silu_kernel, g1, blk, 0, stream, xbf, w13t, hbuf);
  dim3 g2(DMODEL / 128, GTOK / 128, NEXP);  // 16 x 16 x 8
  hipLaunchKernelGGL(moe_gemm2_kernel, g2, blk, 0, stream, hbuf, w2t, out);

  hipLaunchKernelGGL(moe_scores_kernel, dim3(NEXP), blk, 0, stream, ts, out);
}
